// TFLSGCNCNN_49263274885787
// MI455X (gfx1250) — matrix operand order repaired and checked by reading, not compiled, not run on hardware
//
#include <hip/hip_runtime.h>
#include <stdint.h>

// ---------------------------------------------------------------------------
// TFLSGCN-CNN for MI455X (gfx1250, wave32).
// Key algebraic opt: (A@A)@x == A@(A@x)  ->  never materialize A^2.
// Big GEMMs run as split-bf16 (hi+lo) WMMA:  C += Ahi*Bhi + Ahi*Blo + Alo*Bhi
// (3x v_wmma_f32_16x16x32_bf16 per tile, fp32 accumulate, ~2^-16 rel error).
// GEMM main loop: double-buffered LDS; staging via GLOBAL_LOAD_ASYNC_TO_LDS
// (ASYNCcnt path) when the toolchain exposes the builtin, sync fallback else.
// ---------------------------------------------------------------------------

typedef __attribute__((ext_vector_type(16))) __bf16 v16bf;
typedef __attribute__((ext_vector_type(8)))  float  v8f;

#define NNODE 4096
#define BATCH 32
#define DI    32
#define DWIN  16
#define OD2   64
#define DEMB  10
#define NCOL  1024   // BATCH*DI columns of the big GEMMs

union BfPack { uint4 q[2]; v16bf v; };

#if __has_builtin(__builtin_amdgcn_global_load_async_to_lds_b128)
#define HAVE_ASYNC_LDS 1
#else
#define HAVE_ASYNC_LDS 0
#endif

#if HAVE_ASYNC_LDS
// Builtin expects clang-vector pointees:  (v4i addrspace(1)*, v4i addrspace(3)*, Ii, Ii)
typedef int v4i __attribute__((vector_size(4 * sizeof(int))));
typedef __attribute__((address_space(1))) v4i g_v4i;   // global (prints __device__)
typedef __attribute__((address_space(3))) v4i l_v4i;   // LDS    (prints __shared__)
__device__ __forceinline__ void async_b128(const uint4* g, uint4* l)
{
    __builtin_amdgcn_global_load_async_to_lds_b128(
        (g_v4i*)g, (l_v4i*)l, /*offset=*/0, /*cpol=*/0);
}
#endif

__device__ __forceinline__ void wait_async_zero()
{
#if HAVE_ASYNC_LDS
#if __has_builtin(__builtin_amdgcn_s_wait_asynccnt)
    __builtin_amdgcn_s_wait_asynccnt(0);
#else
    asm volatile("s_wait_asynccnt 0x0" ::: "memory");
#endif
#endif
}

// ---------------------------------------------------------------------------
// 1) A = softmax(relu(E @ E^T), axis=1), emitted directly as bf16 hi/lo split.
//    One workgroup per row; row staged in LDS for the two softmax passes.
// ---------------------------------------------------------------------------
__global__ __launch_bounds__(256)
void k_adj(const float* __restrict__ E, __bf16* __restrict__ Ahi,
           __bf16* __restrict__ Alo)
{
    const int n = blockIdx.x;
    __shared__ float srow[NNODE];
    __shared__ float sred[256];
    float e[DEMB];
#pragma unroll
    for (int d = 0; d < DEMB; ++d) e[d] = E[n * DEMB + d];

    float lmax = -3.0e38f;
    for (int m = threadIdx.x; m < NNODE; m += 256) {
        float s = 0.f;
#pragma unroll
        for (int d = 0; d < DEMB; ++d) s += e[d] * E[m * DEMB + d];
        s = fmaxf(s, 0.f);          // relu
        srow[m] = s;
        lmax = fmaxf(lmax, s);
    }
    sred[threadIdx.x] = lmax; __syncthreads();
    for (int off = 128; off > 0; off >>= 1) {
        if (threadIdx.x < off)
            sred[threadIdx.x] = fmaxf(sred[threadIdx.x], sred[threadIdx.x + off]);
        __syncthreads();
    }
    const float rmax = sred[0];
    __syncthreads();

    float lsum = 0.f;
    for (int m = threadIdx.x; m < NNODE; m += 256) {
        float ev = __expf(srow[m] - rmax);
        srow[m] = ev;
        lsum += ev;
    }
    sred[threadIdx.x] = lsum; __syncthreads();
    for (int off = 128; off > 0; off >>= 1) {
        if (threadIdx.x < off)
            sred[threadIdx.x] += sred[threadIdx.x + off];
        __syncthreads();
    }
    const float inv = 1.0f / sred[0];

    for (int m = threadIdx.x; m < NNODE; m += 256) {
        float v = srow[m] * inv;
        __bf16 h = (__bf16)v;
        Ahi[(size_t)n * NNODE + m] = h;
        Alo[(size_t)n * NNODE + m] = (__bf16)(v - (float)h);
    }
}

// ---------------------------------------------------------------------------
// 2) Split/transpose producers for the GEMM B operand: Bt[col j][k] bf16 hi/lo
//    so K-pairs are contiguous (matches the 16-bit B-matrix VGPR layout).
// ---------------------------------------------------------------------------
__global__ __launch_bounds__(256)
void k_split_x(const float* __restrict__ x, __bf16* __restrict__ Bhi,
               __bf16* __restrict__ Blo)
{
    int gid = blockIdx.x * 256 + threadIdx.x;       // NCOL*NNODE threads
    int m = gid & (NNODE - 1);
    int j = gid >> 12;                               // 0..1023
    int b = j >> 5, i = j & 31;
    float v = x[((size_t)b * NNODE + m) * DI + i];
    __bf16 h = (__bf16)v;
    size_t o = (size_t)j * NNODE + m;
    Bhi[o] = h;
    Blo[o] = (__bf16)(v - (float)h);
}

__global__ __launch_bounds__(256)
void k_split_y(const float* __restrict__ y, __bf16* __restrict__ Bhi,
               __bf16* __restrict__ Blo)
{
    int gid = blockIdx.x * 256 + threadIdx.x;
    int m = gid & (NNODE - 1);
    int j = gid >> 12;
    float v = y[(size_t)m * NCOL + j];
    __bf16 h = (__bf16)v;
    size_t o = (size_t)j * NNODE + m;
    Bhi[o] = h;
    Blo[o] = (__bf16)(v - (float)h);
}

// ---------------------------------------------------------------------------
// 3) Split-bf16 GEMM: C[4096 x 1024] = (Ahi+Alo)[4096x4096] * (Bhi+Blo)^T.
//    A row-major [row][k], Bt row-major [col][k]. 128x128 block tile, K-step 32.
//    8 waves; each wave owns a 32x64 region = 2x4 16x16 f32 accumulators.
//    Double-buffered LDS (2 x 32KB): stage slab k+32 while computing slab k,
//    one barrier per K-step. Async staging writes LDS directly (ASYNCcnt).
// ---------------------------------------------------------------------------
__global__ __launch_bounds__(256)
void k_gemm_bf16x3(const __bf16* __restrict__ Ahi, const __bf16* __restrict__ Alo,
                   const __bf16* __restrict__ Bhi, const __bf16* __restrict__ Blo,
                   float* __restrict__ C, int Ncols, int Kdim)
{
    __shared__ uint4 sAh[2][128 * 4];  // [buf][row][4 x uint4] = 16 bf16-pairs/row
    __shared__ uint4 sAl[2][128 * 4];
    __shared__ uint4 sBh[2][128 * 4];  // [buf][col][4 x uint4]
    __shared__ uint4 sBl[2][128 * 4];

    const int tid   = threadIdx.x;
    const int lane  = tid & 31;
    const int wave  = tid >> 5;
    const int waveM = wave >> 1;     // 0..3 -> 32-row slab
    const int waveN = wave & 1;      // 0..1 -> 64-col slab
    const int half  = lane >> 4;
    const int l16   = lane & 15;

    const int rowBase = blockIdx.y * 128;
    const int colBase = blockIdx.x * 128;

    const v8f vzero = {0.f, 0.f, 0.f, 0.f, 0.f, 0.f, 0.f, 0.f};
    v8f acc[2][4];
#pragma unroll
    for (int i = 0; i < 2; ++i)
#pragma unroll
        for (int j = 0; j < 4; ++j) acc[i][j] = vzero;

    const int sr = tid >> 1;         // staging row 0..127
    const int sc = (tid & 1) * 2;    // staging uint4 pair 0 or 2

    // stage one 32-wide K slab into buffer `buf`
    auto stage = [&](int k0, int buf) {
        const uint4* gAh = (const uint4*)(Ahi + (size_t)(rowBase + sr) * Kdim + k0);
        const uint4* gAl = (const uint4*)(Alo + (size_t)(rowBase + sr) * Kdim + k0);
        const uint4* gBh = (const uint4*)(Bhi + (size_t)(colBase + sr) * Kdim + k0);
        const uint4* gBl = (const uint4*)(Blo + (size_t)(colBase + sr) * Kdim + k0);
#if HAVE_ASYNC_LDS
        async_b128(gAh + sc + 0, &sAh[buf][sr * 4 + sc + 0]);
        async_b128(gAh + sc + 1, &sAh[buf][sr * 4 + sc + 1]);
        async_b128(gAl + sc + 0, &sAl[buf][sr * 4 + sc + 0]);
        async_b128(gAl + sc + 1, &sAl[buf][sr * 4 + sc + 1]);
        async_b128(gBh + sc + 0, &sBh[buf][sr * 4 + sc + 0]);
        async_b128(gBh + sc + 1, &sBh[buf][sr * 4 + sc + 1]);
        async_b128(gBl + sc + 0, &sBl[buf][sr * 4 + sc + 0]);
        async_b128(gBl + sc + 1, &sBl[buf][sr * 4 + sc + 1]);
#else
        sAh[buf][sr * 4 + sc + 0] = gAh[sc + 0];
        sAh[buf][sr * 4 + sc + 1] = gAh[sc + 1];
        sAl[buf][sr * 4 + sc + 0] = gAl[sc + 0];
        sAl[buf][sr * 4 + sc + 1] = gAl[sc + 1];
        sBh[buf][sr * 4 + sc + 0] = gBh[sc + 0];
        sBh[buf][sr * 4 + sc + 1] = gBh[sc + 1];
        sBl[buf][sr * 4 + sc + 0] = gBl[sc + 0];
        sBl[buf][sr * 4 + sc + 1] = gBl[sc + 1];
#endif
        if (k0 + 32 < Kdim) {        // global_prefetch_b8 the slab after next
            __builtin_prefetch((const char*)gAh + 64, 0, 3);
            __builtin_prefetch((const char*)gAl + 64, 0, 3);
            __builtin_prefetch((const char*)gBh + 64, 0, 3);
            __builtin_prefetch((const char*)gBl + 64, 0, 3);
        }
    };

    stage(0, 0);
    wait_async_zero();
    __syncthreads();

    for (int k0 = 0; k0 < Kdim; k0 += 32) {
        const int cur = (k0 >> 5) & 1;
        if (k0 + 32 < Kdim) stage(k0 + 32, cur ^ 1);

        // A operand per ISA 16x32 bf16 layout: lane(m,half); VGPR v: pair
        // g*8 + half*4 + r  ->  uint4 indices {half, 2+half}
        BfPack aH[2], aL[2];
#pragma unroll
        for (int sm = 0; sm < 2; ++sm) {
            int m = waveM * 32 + sm * 16 + l16;
            aH[sm].q[0] = sAh[cur][m * 4 + half];
            aH[sm].q[1] = sAh[cur][m * 4 + 2 + half];
            aL[sm].q[0] = sAl[cur][m * 4 + half];
            aL[sm].q[1] = sAl[cur][m * 4 + 2 + half];
        }
#pragma unroll
        for (int sn = 0; sn < 4; ++sn) {
            // B operand 32x16: lane(n,half); same K order as A
            //   -> uint4 indices {half, 2+half}
            int nn = waveN * 64 + sn * 16 + l16;
            BfPack bH, bL;
            bH.q[0] = sBh[cur][nn * 4 + half];
            bH.q[1] = sBh[cur][nn * 4 + 2 + half];
            bL.q[0] = sBl[cur][nn * 4 + half];
            bL.q[1] = sBl[cur][nn * 4 + 2 + half];
#pragma unroll
            for (int sm = 0; sm < 2; ++sm) {
                acc[sm][sn] = __builtin_amdgcn_wmma_f32_16x16x32_bf16(
                    false, aH[sm].v, false, bH.v, (short)0, acc[sm][sn], false, false);
                acc[sm][sn] = __builtin_amdgcn_wmma_f32_16x16x32_bf16(
                    false, aH[sm].v, false, bL.v, (short)0, acc[sm][sn], false, false);
                acc[sm][sn] = __builtin_amdgcn_wmma_f32_16x16x32_bf16(
                    false, aL[sm].v, false, bH.v, (short)0, acc[sm][sn], false, false);
            }
        }

        // own async staging done -> barrier makes next buffer visible to all
        // waves and guarantees everyone finished reading buffer `cur`.
        wait_async_zero();
        __syncthreads();
    }

    // C/D 16x16 f32 layout: VGPR v -> M = v + 8*half, N = lane&15
#pragma unroll
    for (int sm = 0; sm < 2; ++sm)
#pragma unroll
        for (int sn = 0; sn < 4; ++sn) {
            int col = colBase + waveN * 64 + sn * 16 + l16;
#pragma unroll
            for (int v = 0; v < 8; ++v) {
                int row = rowBase + waveM * 32 + sm * 16 + half * 8 + v;
                C[(size_t)row * Ncols + col] = acc[sm][sn][v];
            }
        }
}

// ---------------------------------------------------------------------------
// 4) Topo CNN: fused conv5x5+relu+maxpool2 (x2), then FC reduction.
// ---------------------------------------------------------------------------
__global__ __launch_bounds__(256)
void k_conv1(const float* __restrict__ pi, const float* __restrict__ w1,
             const float* __restrict__ b1, float* __restrict__ h1)
{
    int gid = blockIdx.x * 256 + threadIdx.x;   // 32*16*50*50
    if (gid >= 32 * 16 * 2500) return;
    int xo = gid % 50, yo = (gid / 50) % 50, c = (gid / 2500) % 16, b = gid / 40000;
    float w[25];
#pragma unroll
    for (int k = 0; k < 25; ++k) w[k] = w1[c * 25 + k];
    const float bias = b1[c];
    const float* pb = pi + (size_t)b * 10000;
    float mx = -3.0e38f;
#pragma unroll
    for (int dy = 0; dy < 2; ++dy)
#pragma unroll
        for (int dx = 0; dx < 2; ++dx) {
            int yy = yo * 2 + dy, xx = xo * 2 + dx;
            float s = bias;
#pragma unroll
            for (int ky = 0; ky < 5; ++ky) {
                int iy = yy + ky - 2;
                if (iy < 0 || iy >= 100) continue;
#pragma unroll
                for (int kx = 0; kx < 5; ++kx) {
                    int ix = xx + kx - 2;
                    if (ix < 0 || ix >= 100) continue;
                    s += w[ky * 5 + kx] * pb[iy * 100 + ix];
                }
            }
            mx = fmaxf(mx, fmaxf(s, 0.f));
        }
    h1[gid] = mx;
}

__global__ __launch_bounds__(256)
void k_conv2(const float* __restrict__ h1, const float* __restrict__ w2,
             const float* __restrict__ b2, float* __restrict__ h2)
{
    int b = blockIdx.x >> 5, oc = blockIdx.x & 31;
    __shared__ float sw[400];
    for (int i = threadIdx.x; i < 400; i += 256) sw[i] = w2[oc * 400 + i];
    __syncthreads();
    const float bias = b2[oc];
    for (int p = threadIdx.x; p < 625; p += 256) {
        int xo = p % 25, yo = p / 25;
        float mx = -3.0e38f;
        for (int dy = 0; dy < 2; ++dy)
            for (int dx = 0; dx < 2; ++dx) {
                int yy = yo * 2 + dy, xx = xo * 2 + dx;
                float s = bias;
                for (int ic = 0; ic < 16; ++ic) {
                    const float* hp = h1 + ((size_t)b * 16 + ic) * 2500;
                    const float* wp = sw + ic * 25;
#pragma unroll
                    for (int ky = 0; ky < 5; ++ky) {
                        int iy = yy + ky - 2;
                        if (iy < 0 || iy >= 50) continue;
#pragma unroll
                        for (int kx = 0; kx < 5; ++kx) {
                            int ix = xx + kx - 2;
                            if (ix < 0 || ix >= 50) continue;
                            s += wp[ky * 5 + kx] * hp[iy * 50 + ix];
                        }
                    }
                }
                mx = fmaxf(mx, fmaxf(s, 0.f));
            }
        h2[((size_t)(b * 32 + oc)) * 625 + p] = mx;
    }
}

__global__ __launch_bounds__(256)
void k_fc(const float* __restrict__ h2, const float* __restrict__ fw,
          const float* __restrict__ fb, float* __restrict__ topo)
{
    int b = blockIdx.x >> 6, o = blockIdx.x & 63;
    const float* hv = h2 + (size_t)b * 20000;
    const float* wv = fw + (size_t)o * 20000;
    float s = 0.f;
    for (int i = threadIdx.x; i < 20000; i += 256) s += hv[i] * wv[i];
    __shared__ float sr[256];
    sr[threadIdx.x] = s; __syncthreads();
    for (int off = 128; off > 0; off >>= 1) {
        if (threadIdx.x < off) sr[threadIdx.x] += sr[threadIdx.x + off];
        __syncthreads();
    }
    if (threadIdx.x == 0) topo[blockIdx.x] = sr[0] + fb[o];
}

// ---------------------------------------------------------------------------
// 5) Fused per-node epilogue: weight generation from pools, grouped gconv,
//    temporal window conv, topo gating, bias, concat. One workgroup per node.
// ---------------------------------------------------------------------------
__global__ __launch_bounds__(256)
void k_node(const float* __restrict__ x, const float* __restrict__ y1,
            const float* __restrict__ y2, const float* __restrict__ xwin,
            const float* __restrict__ E, const float* __restrict__ wpool,
            const float* __restrict__ wwin, const float* __restrict__ bpool,
            const float* __restrict__ Tw, const float* __restrict__ topo,
            float* __restrict__ out)
{
    const int n = blockIdx.x;
    __shared__ float sW[96 * 64];       // node weights  [k*32+i][o]
    __shared__ float sWW[16 * 64];      // window weights [i][o]
    __shared__ float sBias[128];
    __shared__ float sXg[3 * 1024];     // [k][b*32+i]
    __shared__ float sXw[32 * 16];      // [b][i]
    __shared__ float sE[DEMB];
    const int t = threadIdx.x;
    if (t < DEMB) sE[t] = E[n * DEMB + t];
    __syncthreads();
    float e[DEMB];
#pragma unroll
    for (int d = 0; d < DEMB; ++d) e[d] = sE[d];

    for (int idx = t; idx < 6144; idx += 256) {
        float s = 0.f;
#pragma unroll
        for (int d = 0; d < DEMB; ++d) s += e[d] * wpool[d * 6144 + idx];
        sW[idx] = s;
    }
    for (int idx = t; idx < 1024; idx += 256) {
        float s = 0.f;
#pragma unroll
        for (int d = 0; d < DEMB; ++d) s += e[d] * wwin[d * 1024 + idx];
        sWW[idx] = s;
    }
    if (t < 128) {
        float s = 0.f;
#pragma unroll
        for (int d = 0; d < DEMB; ++d) s += e[d] * bpool[d * 128 + t];
        sBias[t] = s;
    }
    for (int idx = t; idx < 1024; idx += 256) {
        int b = idx >> 5, i = idx & 31;
        sXg[0 * 1024 + idx] = x[((size_t)b * NNODE + n) * DI + i];
        sXg[1 * 1024 + idx] = y1[(size_t)n * NCOL + idx];
        sXg[2 * 1024 + idx] = y2[(size_t)n * NCOL + idx];
    }
    for (int idx = t; idx < 512; idx += 256) {
        int b = idx >> 4, i = idx & 15;
        float s = 0.f;
#pragma unroll
        for (int tt = 0; tt < 12; ++tt)
            s += Tw[tt] * xwin[(((size_t)b * 12 + tt) * NNODE + n) * DWIN + i];
        sXw[idx] = s;
    }
    __syncthreads();

    for (int idx = t; idx < 2048; idx += 256) {
        int b = idx >> 6, o = idx & 63;
        float g = 0.f;
#pragma unroll
        for (int k = 0; k < 3; ++k)
            for (int i = 0; i < 32; ++i)
                g += sXg[k * 1024 + b * 32 + i] * sW[(k * 32 + i) * 64 + o];
        float w = 0.f;
#pragma unroll
        for (int i = 0; i < 16; ++i) w += sXw[b * 16 + i] * sWW[i * 64 + o];
        float tg = topo[b * 64 + o];
        size_t base = ((size_t)b * NNODE + n) * 128;
        out[base + o]      = g * tg + sBias[o];
        out[base + 64 + o] = w * tg + sBias[64 + o];
    }
}

// ---------------------------------------------------------------------------
extern "C" void kernel_launch(void* const* d_in, const int* in_sizes, int n_in,
                              void* d_out, int out_size, void* d_ws, size_t ws_size,
                              hipStream_t stream)
{
    (void)in_sizes; (void)n_in; (void)out_size; (void)ws_size;
    const float* x     = (const float*)d_in[0];
    const float* xwin  = (const float*)d_in[1];
    const float* E     = (const float*)d_in[2];
    const float* pi    = (const float*)d_in[3];
    const float* wpool = (const float*)d_in[4];
    const float* wwin  = (const float*)d_in[5];
    const float* bpool = (const float*)d_in[6];
    const float* Tw    = (const float*)d_in[7];
    const float* w1    = (const float*)d_in[8];
    const float* b1    = (const float*)d_in[9];
    const float* w2    = (const float*)d_in[10];
    const float* b2    = (const float*)d_in[11];
    const float* fw    = (const float*)d_in[12];
    const float* fb    = (const float*)d_in[13];
    float* out = (float*)d_out;

    char* ws = (char*)d_ws;
    size_t off = 0;
    auto take = [&](size_t bytes) -> void* {
        off = (off + 255) & ~(size_t)255;
        void* p = ws + off;
        off += bytes;
        return p;
    };
    const size_t NN = (size_t)NNODE * NNODE;
    __bf16* Ahi = (__bf16*)take(NN * 2);
    __bf16* Alo = (__bf16*)take(NN * 2);
    __bf16* Xhi = (__bf16*)take((size_t)NCOL * NNODE * 2);
    __bf16* Xlo = (__bf16*)take((size_t)NCOL * NNODE * 2);
    float*  y1  = (float*)take((size_t)NNODE * NCOL * 4);
    __bf16* Yhi = (__bf16*)take((size_t)NCOL * NNODE * 2);
    __bf16* Ylo = (__bf16*)take((size_t)NCOL * NNODE * 2);
    float*  y2  = (float*)take((size_t)NNODE * NCOL * 4);
    float*  h1  = (float*)take((size_t)32 * 16 * 2500 * 4);
    float*  h2  = (float*)take((size_t)32 * 32 * 625 * 4);
    float*  topo = (float*)take((size_t)32 * 64 * 4);

    // adjacency + operand preparation
    k_adj<<<NNODE, 256, 0, stream>>>(E, Ahi, Alo);
    k_split_x<<<(NNODE * NCOL) / 256, 256, 0, stream>>>(x, Xhi, Xlo);

    // topo CNN (independent chain)
    k_conv1<<<(32 * 16 * 2500 + 255) / 256, 256, 0, stream>>>(pi, w1, b1, h1);
    k_conv2<<<32 * 32, 256, 0, stream>>>(h1, w2, b2, h2);
    k_fc<<<32 * 64, 256, 0, stream>>>(h2, fw, fb, topo);

    // y1 = A*x ; y2 = A*y1   (replaces materializing A@A)
    k_gemm_bf16x3<<<dim3(NCOL / 128, NNODE / 128), 256, 0, stream>>>(
        Ahi, Alo, Xhi, Xlo, y1, NCOL, NNODE);
    k_split_y<<<(NNODE * NCOL) / 256, 256, 0, stream>>>(y1, Yhi, Ylo);
    k_gemm_bf16x3<<<dim3(NCOL / 128, NNODE / 128), 256, 0, stream>>>(
        Ahi, Alo, Yhi, Ylo, y2, NCOL, NNODE);

    // fused per-node epilogue -> out [B, N, 128]
    k_node<<<NNODE, 256, 0, stream>>>(x, y1, y2, xwin, E, wpool, wwin, bpool,
                                      Tw, topo, out);
}
